// DABCConv2d_41609643163689
// MI455X (gfx1250) — 8-bit matrix operand order repaired and checked by reading, compiled, not run on hardware
//
#include <hip/hip_runtime.h>
#include <hip/hip_bf16.h>
#include <stdint.h>

typedef int   v8i  __attribute__((ext_vector_type(8)));
typedef int   v16i __attribute__((ext_vector_type(16)));
typedef float v8f  __attribute__((ext_vector_type(8)));
typedef unsigned int u32;

#define BATCH 8
#define CH    64
#define HH    256
#define WW    256
#define HWN   65536
#define EPSV  1e-6f

#define FP8_P1 ((uint8_t)0x38)
#define FP8_M1 ((uint8_t)0xB8)
#define FP8_Z  ((uint8_t)0x00)

__global__ void __launch_bounds__(256)
stats_kernel(const float* __restrict__ x, const float* __restrict__ b_move,
             float* __restrict__ stats) {
  int bc = blockIdx.x;
  int c  = bc & 63;
  float bias = b_move[c];
  const float4* xp = (const float4*)(x + (size_t)bc * HWN);
  float s = 0.f, s2 = 0.f, sa = 0.f;
  for (int i = threadIdx.x; i < HWN / 4; i += 256) {
    float4 v = xp[i];
    float a0 = v.x + bias, a1 = v.y + bias, a2 = v.z + bias, a3 = v.w + bias;
    s  += a0 + a1 + a2 + a3;
    s2 += a0*a0 + a1*a1 + a2*a2 + a3*a3;
    sa += fabsf(a0) + fabsf(a1) + fabsf(a2) + fabsf(a3);
  }
  __shared__ float r0[256], r1[256], r2[256];
  r0[threadIdx.x] = s; r1[threadIdx.x] = s2; r2[threadIdx.x] = sa;
  __syncthreads();
  for (int off = 128; off > 0; off >>= 1) {
    if ((int)threadIdx.x < off) {
      r0[threadIdx.x] += r0[threadIdx.x + off];
      r1[threadIdx.x] += r1[threadIdx.x + off];
      r2[threadIdx.x] += r2[threadIdx.x + off];
    }
    __syncthreads();
  }
  if (threadIdx.x == 0) {
    float sum = r0[0], sum2 = r1[0], sabs = r2[0];
    float mean = sum * (1.f / HWN);
    float var  = (sum2 - sum * mean) * (1.f / (HWN - 1)) + EPSV;
    stats[0 * 512 + bc] = sabs * (1.f / HWN);
    stats[1 * 512 + bc] = mean;
    stats[2 * 512 + bc] = sqrtf(var);
  }
}

__global__ void __launch_bounds__(256)
binarize_kernel(const float* __restrict__ x, const float* __restrict__ b_move,
                uint8_t* __restrict__ act) {
  int bid = blockIdx.x;
  int wb = bid & 3;
  int h  = (bid >> 2) & 255;
  int b  = bid >> 10;
  int w0 = wb * 64;
  __shared__ __attribute__((aligned(16))) uint8_t tile[64 * 64];
  int tid = threadIdx.x;
  for (int i = tid; i < 4096; i += 256) {
    int c = i >> 6, w = i & 63;
    float v = x[(((size_t)b * CH + c) * HH + h) * WW + (w0 + w)] + b_move[c];
    tile[w * 64 + c] = (v > 0.f) ? FP8_P1 : ((v < 0.f) ? FP8_M1 : FP8_Z);
  }
  __syncthreads();
  u32* dst = (u32*)(act + ((((size_t)b * HH + h) * WW) + w0) * 64);
  const u32* src = (const u32*)tile;
  for (int i = tid; i < 1024; i += 256) dst[i] = src[i];
}

__global__ void __launch_bounds__(256)
prep_kernel(const float* __restrict__ w_conv, const float* __restrict__ w1d,
            const float* __restrict__ stats,
            float* __restrict__ dscale, float* __restrict__ wscale,
            uint8_t* __restrict__ wpack) {
  int tid = threadIdx.x;
  if (tid < 64) {
    float s = 0.f;
    for (int i = 0; i < 576; ++i) s += fabsf(w_conv[tid * 576 + i]);
    wscale[tid] = s * (1.f / 576.f);
  }
  for (int i = tid; i < 512; i += 256) {
    int c = i & 63;
    int base = i - c;
    float y = 0.f;
    for (int f = 0; f < 3; ++f)
      for (int j = 0; j < 3; ++j) {
        int cc = c + j - 1;
        if (cc >= 0 && cc < 64) y += w1d[f * 3 + j] * stats[f * 512 + base + cc];
      }
    dscale[i] = 1.f / (1.f + __expf(-y));
  }
  for (int idx = tid; idx < 64 * 64 * 9; idx += 256) {
    int kw = idx % 3, t1 = idx / 3;
    int kh = t1 % 3,  t2 = t1 / 3;
    int ci = t2 & 63, co = t2 >> 6;
    float wv = w_conv[idx];
    uint8_t sg = (wv > 0.f) ? FP8_P1 : ((wv < 0.f) ? FP8_M1 : FP8_Z);
    int tap = kh * 3 + kw;
    int ct = co >> 4, m = co & 15;
    int hi = (ci >> 4) & 1;
    int lane = m + hi * 16;
    int g = ci >> 5;
    int o = ci & 15;
    int byteidx = g * 16 + o;
    wpack[(((size_t)(tap * 4 + ct) * 32 + lane) * 32) + byteidx] = sg;
  }
}

__global__ void __launch_bounds__(256)
conv_kernel(const uint8_t* __restrict__ act, const u32* __restrict__ wpack,
            const float* __restrict__ x,
            const float* __restrict__ wscale, const float* __restrict__ dscale,
            const float* __restrict__ pr_b0, const float* __restrict__ prelu_a,
            const float* __restrict__ pr_b1,
            float* __restrict__ out) {
  __shared__ __attribute__((aligned(16))) u32 wlds[9216];
  __shared__ __attribute__((aligned(16))) u32 alds[3168];
  __shared__ float plds[320];

  int bid = blockIdx.x;
  int wb = bid & 3, h = (bid >> 2) & 255, b = bid >> 10;
  int w0 = wb * 64;
  int tid = threadIdx.x;

  for (int i = tid; i < 9216; i += 256) wlds[i] = wpack[i];
  if (tid < 64) {
    plds[tid]       = wscale[tid];
    plds[64 + tid]  = dscale[b * 64 + tid];
    plds[128 + tid] = pr_b0[tid];
    plds[192 + tid] = prelu_a[tid];
    plds[256 + tid] = pr_b1[tid];
  }
  const u32* actd = (const u32*)act;
  for (int d = tid; d < 3168; d += 256) {
    int r = d / 1056, rr = d % 1056;
    int px = rr >> 4, dw = rr & 15;
    int gh = h - 1 + r, gw = w0 - 1 + px;
    u32 v = 0u;
    if (gh >= 0 && gh < HH && gw >= 0 && gw < WW)
      v = actd[(((size_t)b * HH + gh) * WW + gw) * 16 + dw];
    alds[d] = v;
  }
  __syncthreads();

  int wv = tid >> 5, lane = tid & 31;
  int ct = wv & 3;
  int pb = wv >> 2;
  int n = lane & 15, khi = lane >> 4;

  const uint4* wl4 = (const uint4*)wlds;
  const uint4* al4 = (const uint4*)alds;

  v8f acc0 = {0.f, 0.f, 0.f, 0.f, 0.f, 0.f, 0.f, 0.f};
  v8f acc1 = {0.f, 0.f, 0.f, 0.f, 0.f, 0.f, 0.f, 0.f};

#pragma unroll
  for (int kh = 0; kh < 3; ++kh) {
    int t0 = kh * 3;
    int ai0 = ((t0 * 4 + ct) * 32 + lane) * 2;
    int ai1 = (((t0 + 1) * 4 + ct) * 32 + lane) * 2;
    int ai2 = (((t0 + 2) * 4 + ct) * 32 + lane) * 2;
    uint4 wa = wl4[ai0],  wb_ = wl4[ai0 + 1];
    uint4 wc = wl4[ai1],  wd = wl4[ai1 + 1];
    v16i A01;
    A01[0]  = (int)wa.x;  A01[1]  = (int)wa.y;  A01[2]  = (int)wa.z;  A01[3]  = (int)wa.w;
    A01[4]  = (int)wb_.x; A01[5]  = (int)wb_.y; A01[6]  = (int)wb_.z; A01[7]  = (int)wb_.w;
    A01[8]  = (int)wc.x;  A01[9]  = (int)wc.y;  A01[10] = (int)wc.z;  A01[11] = (int)wc.w;
    A01[12] = (int)wd.x;  A01[13] = (int)wd.y;  A01[14] = (int)wd.z;  A01[15] = (int)wd.w;
    uint4 we = wl4[ai2],  wf = wl4[ai2 + 1];
    v8i A2;
    A2[0] = (int)we.x; A2[1] = (int)we.y; A2[2] = (int)we.z; A2[3] = (int)we.w;
    A2[4] = (int)wf.x; A2[5] = (int)wf.y; A2[6] = (int)wf.z; A2[7] = (int)wf.w;

    {
      int px = pb * 16 + n;
      int bi = kh * 264 + px * 4 + khi;
      uint4 c0 = al4[bi],     c1 = al4[bi + 2];
      uint4 c2 = al4[bi + 4], c3 = al4[bi + 6];
      v16i B01;
      B01[0]  = (int)c0.x; B01[1]  = (int)c0.y; B01[2]  = (int)c0.z; B01[3]  = (int)c0.w;
      B01[4]  = (int)c1.x; B01[5]  = (int)c1.y; B01[6]  = (int)c1.z; B01[7]  = (int)c1.w;
      B01[8]  = (int)c2.x; B01[9]  = (int)c2.y; B01[10] = (int)c2.z; B01[11] = (int)c2.w;
      B01[12] = (int)c3.x; B01[13] = (int)c3.y; B01[14] = (int)c3.z; B01[15] = (int)c3.w;
      acc0 = __builtin_amdgcn_wmma_f32_16x16x128_fp8_fp8(A01, B01, (short)0, acc0,
                                                         false, false);
      uint4 c4 = al4[bi + 8], c5 = al4[bi + 10];
      v8i B2;
      B2[0] = (int)c4.x; B2[1] = (int)c4.y; B2[2] = (int)c4.z; B2[3] = (int)c4.w;
      B2[4] = (int)c5.x; B2[5] = (int)c5.y; B2[6] = (int)c5.z; B2[7] = (int)c5.w;
      acc0 = __builtin_amdgcn_wmma_f32_16x16x64_fp8_fp8(A2, B2, (short)0, acc0,
                                                        false, false);
    }
    {
      int px = (pb + 2) * 16 + n;
      int bi = kh * 264 + px * 4 + khi;
      uint4 c0 = al4[bi],     c1 = al4[bi + 2];
      uint4 c2 = al4[bi + 4], c3 = al4[bi + 6];
      v16i B01;
      B01[0]  = (int)c0.x; B01[1]  = (int)c0.y; B01[2]  = (int)c0.z; B01[3]  = (int)c0.w;
      B01[4]  = (int)c1.x; B01[5]  = (int)c1.y; B01[6]  = (int)c1.z; B01[7]  = (int)c1.w;
      B01[8]  = (int)c2.x; B01[9]  = (int)c2.y; B01[10] = (int)c2.z; B01[11] = (int)c2.w;
      B01[12] = (int)c3.x; B01[13] = (int)c3.y; B01[14] = (int)c3.z; B01[15] = (int)c3.w;
      acc1 = __builtin_amdgcn_wmma_f32_16x16x128_fp8_fp8(A01, B01, (short)0, acc1,
                                                         false, false);
      uint4 c4 = al4[bi + 8], c5 = al4[bi + 10];
      v8i B2;
      B2[0] = (int)c4.x; B2[1] = (int)c4.y; B2[2] = (int)c4.z; B2[3] = (int)c4.w;
      B2[4] = (int)c5.x; B2[5] = (int)c5.y; B2[6] = (int)c5.z; B2[7] = (int)c5.w;
      acc1 = __builtin_amdgcn_wmma_f32_16x16x64_fp8_fp8(A2, B2, (short)0, acc1,
                                                        false, false);
    }
  }

#pragma unroll
  for (int v = 0; v < 8; ++v) {
    int co = ct * 16 + v + khi * 8;
    float ws_ = plds[co], ds_ = plds[64 + co];
    float b0_ = plds[128 + co], a_ = plds[192 + co], b1_ = plds[256 + co];
    {
      int w = w0 + pb * 16 + n;
      size_t oidx = (((size_t)b * CH + co) * HH + h) * WW + w;
      float f = acc0[v] * ws_ * ds_ + b0_;
      f = (f > 0.f) ? f : a_ * f;
      out[oidx] = f + b1_ + x[oidx];
    }
    {
      int w = w0 + (pb + 2) * 16 + n;
      size_t oidx = (((size_t)b * CH + co) * HH + h) * WW + w;
      float f = acc1[v] * ws_ * ds_ + b0_;
      f = (f > 0.f) ? f : a_ * f;
      out[oidx] = f + b1_ + x[oidx];
    }
  }
}

extern "C" void kernel_launch(void* const* d_in, const int* in_sizes, int n_in,
                              void* d_out, int out_size, void* d_ws, size_t ws_size,
                              hipStream_t stream) {
  const float* x     = (const float*)d_in[0];
  const float* bmove = (const float*)d_in[1];
  const float* wconv = (const float*)d_in[2];
  const float* w1d   = (const float*)d_in[3];
  const float* prb0  = (const float*)d_in[4];
  const float* pra   = (const float*)d_in[5];
  const float* prb1  = (const float*)d_in[6];
  float* out = (float*)d_out;

  uint8_t* ws    = (uint8_t*)d_ws;
  uint8_t* act   = ws;
  float*   stats = (float*)(ws + (size_t)BATCH * HH * WW * CH);
  float*   dscal = stats + 3 * 512;
  float*   wscal = dscal + 512;
  uint8_t* wpack = (uint8_t*)(wscal + 64);

  stats_kernel   <<<512,  256, 0, stream>>>(x, bmove, stats);
  binarize_kernel<<<8192, 256, 0, stream>>>(x, bmove, act);
  prep_kernel    <<<1,    256, 0, stream>>>(wconv, w1d, stats, dscal, wscal, wpack);
  conv_kernel    <<<8192, 256, 0, stream>>>(act, (const u32*)wpack, x,
                                            wscal, dscal, prb0, pra, prb1, out);
}
